// cvx_86234353369507
// MI455X (gfx1250) — compile-verified, matrix operand order checked by reading, not run on hardware
//
#include <hip/hip_runtime.h>
#include <hip/hip_bf16.h>

// ---------------------------------------------------------------------------
// Problem constants (fixed by the reference)
// ---------------------------------------------------------------------------
#define NN      4096      // nodes
#define EE      8192      // edges
#define DIN     32
#define HF      256       // hidden / latent feature dim (H == L == 256)
#define NT      1024      // threads in the persistent solve workgroup
#define EPT     (EE/NT)   // 8 edges per thread
#define NPT     (NN/NT)   // 4 nodes per thread
#define NITERS  200
#define RHO_C   10.0f
#define LR_C    5e-4f
#define EPS_C   1e-9f

typedef __attribute__((ext_vector_type(16))) _Float16 v16h;
typedef __attribute__((ext_vector_type(8)))  _Float16 v8h;
typedef __attribute__((ext_vector_type(8)))  float    v8f;

__device__ __forceinline__ float sgnf(float x) {
    return (x > 0.f ? 1.f : (x < 0.f ? -1.f : 0.f));
}
__device__ __forceinline__ float sigmoidf(float x) {
    return 1.f / (1.f + expf(-x));
}

// ---------------------------------------------------------------------------
// Small utility kernels
// ---------------------------------------------------------------------------
__global__ void k_f32_to_f16(const float* __restrict__ in, _Float16* __restrict__ out, int n) {
    int i = blockIdx.x * blockDim.x + threadIdx.x;
    if (i < n) out[i] = (_Float16)in[i];
}

// W [K x Ncols] f32  ->  Wt [Ncols x K] f16  (so B columns are contiguous)
__global__ void k_transpose_f16(const float* __restrict__ W, _Float16* __restrict__ Wt,
                                int K, int Ncols) {
    int i = blockIdx.x * blockDim.x + threadIdx.x;
    if (i < K * Ncols) {
        int k = i / Ncols, n = i % Ncols;
        Wt[(size_t)n * K + k] = (_Float16)W[i];
    }
}

__global__ void k_zero(float* __restrict__ p, int n) {
    int i = blockIdx.x * blockDim.x + threadIdx.x;
    if (i < n) p[i] = 0.f;
}

__global__ void k_deg_scatter(const int* __restrict__ dst, float* __restrict__ deg, int E) {
    int e = blockIdx.x * blockDim.x + threadIdx.x;
    if (e < E) atomicAdd(&deg[dst[e]], 1.f);
}

__global__ void k_deg_fin(const float* __restrict__ deg, float* __restrict__ dis, int n) {
    int i = blockIdx.x * blockDim.x + threadIdx.x;
    if (i < n) dis[i] = rsqrtf(deg[i] + 1.f);
}

// h = relu(in + b[f]); write f32 and f16 copies
__global__ void k_finalize(const float* __restrict__ in, const float* __restrict__ b,
                           float* __restrict__ h32, _Float16* __restrict__ h16, int total) {
    int i = blockIdx.x * blockDim.x + threadIdx.x;
    if (i < total) {
        float v = in[i] + b[i & (HF - 1)];
        v = fmaxf(v, 0.f);
        h32[i] = v;
        h16[i] = (_Float16)v;
    }
}

// agg = t * dis[n]^2   (GCN self-loop term)
__global__ void k_agg_init(const float* __restrict__ t, const float* __restrict__ dis,
                           float* __restrict__ agg, int total) {
    int i = blockIdx.x * blockDim.x + threadIdx.x;
    if (i < total) {
        float d = dis[i >> 8];  // HF == 256
        agg[i] = t[i] * d * d;
    }
}

// per-edge feature scatter: agg[dst] += t[src] * dis[src]*dis[dst]
__global__ void k_edge_scatter(const int* __restrict__ src, const int* __restrict__ dst,
                               const float* __restrict__ dis, const float* __restrict__ t,
                               float* __restrict__ agg) {
    int e = blockIdx.x;
    int f = threadIdx.x;        // blockDim == HF == 256
    int s = src[e], d = dst[e];
    float nrm = dis[s] * dis[d];
    atomicAdd(&agg[(size_t)d * HF + f], t[(size_t)s * HF + f] * nrm);
}

// ---------------------------------------------------------------------------
// WMMA GEMM: C[M x Ncols] = A[M x K](f16) * B (given as Bt[Ncols x K] f16)
// One wave per 16x16 output tile, 4 waves per block, f32 accumulation.
// ---------------------------------------------------------------------------
__global__ __launch_bounds__(128) void k_gemm_f16(
    const _Float16* __restrict__ A, const _Float16* __restrict__ Bt,
    float* __restrict__ C, int M, int Ncols, int K) {
    const int lane  = threadIdx.x & 31;
    const int wave  = threadIdx.x >> 5;
    const int tilesN = Ncols >> 4;
    const int tile   = blockIdx.x * 4 + wave;
    if (tile >= (M >> 4) * tilesN) return;          // wave-uniform guard
    const int tm = tile / tilesN, tn = tile % tilesN;
    const int hsel = lane >> 4;                     // lane half-select (K offset 0/8)
    const int l15  = lane & 15;
    const size_t arow = (size_t)(tm * 16 + l15) * K;
    const size_t brow = (size_t)(tn * 16 + l15) * K;

    v8f acc = {};
    for (int kk = 0; kk < K; kk += 32) {
        const int kb = kk + hsel * 8;
        v8h a0 = *(const v8h*)(A + arow + kb);
        v8h a1 = *(const v8h*)(A + arow + kb + 16);
        v8h b0 = *(const v8h*)(Bt + brow + kb);
        v8h b1 = *(const v8h*)(Bt + brow + kb + 16);
        v16h av = __builtin_shufflevector(a0, a1, 0,1,2,3,4,5,6,7,8,9,10,11,12,13,14,15);
        v16h bv = __builtin_shufflevector(b0, b1, 0,1,2,3,4,5,6,7,8,9,10,11,12,13,14,15);
        acc = __builtin_amdgcn_wmma_f32_16x16x32_f16(
            /*neg_a=*/false, av, /*neg_b=*/false, bv,
            /*c_mod=*/(short)0, acc, /*reuse_a=*/false, /*reuse_b=*/false);
    }
#pragma unroll
    for (int r = 0; r < 8; ++r) {
        int rr = tm * 16 + r + hsel * 8;
        C[(size_t)rr * Ncols + (tn * 16 + l15)] = acc[r];
    }
}

// ---------------------------------------------------------------------------
// Heads: per-node dot products with W_sw (split), W_v; produce s1,s2,v_warm
// ---------------------------------------------------------------------------
__global__ void k_heads(const float* __restrict__ h, const float* __restrict__ Wsw,
                        const float* __restrict__ Wv, const float* __restrict__ bv,
                        float* __restrict__ s1, float* __restrict__ s2,
                        float* __restrict__ vwarm) {
    int lane = threadIdx.x & 31;
    int wave = threadIdx.x >> 5;
    int n = blockIdx.x * 8 + wave;                  // blockDim == 256 -> 8 waves
    float a1 = 0.f, a2 = 0.f, a3 = 0.f;
    for (int f = lane; f < HF; f += 32) {
        float hv = h[(size_t)n * HF + f];
        a1 += hv * Wsw[f];
        a2 += hv * Wsw[HF + f];
        a3 += hv * Wv[f];
    }
#pragma unroll
    for (int o = 16; o > 0; o >>= 1) {
        a1 += __shfl_down(a1, o, 32);
        a2 += __shfl_down(a2, o, 32);
        a3 += __shfl_down(a3, o, 32);
    }
    if (lane == 0) {
        s1[n] = a1;
        s2[n] = a2;
        float vr = sigmoidf(a3 + bv[0]);
        float vw = 0.9f + 0.2f * vr;
        vwarm[n] = vw * vw;
    }
}

__global__ void k_ywarm(const int* __restrict__ src, const int* __restrict__ dst,
                        const float* __restrict__ s1, const float* __restrict__ s2,
                        const float* __restrict__ bsw, float* __restrict__ yw, int E) {
    int e = blockIdx.x * blockDim.x + threadIdx.x;
    if (e < E) yw[e] = sigmoidf(s1[src[e]] + s2[dst[e]] + bsw[0]);
}

// ---------------------------------------------------------------------------
// Persistent single-WGP SOCP penalty solve: 200 iterations, state in LDS+regs.
// Analytic gradient of the penalty objective in the reference.
// ---------------------------------------------------------------------------
__global__ __launch_bounds__(NT) void k_solve(
    const int* __restrict__ src, const int* __restrict__ dst,
    const float* __restrict__ p_inj, const float* __restrict__ q_inj,
    const float* __restrict__ y0,   const float* __restrict__ r_pu,
    const float* __restrict__ x_pu, const float* __restrict__ mflow,
    const float* __restrict__ mv_ptr, const float* __restrict__ sub,
    const float* __restrict__ nsub, const float* __restrict__ mfsq,
    const float* __restrict__ zlsq, const float* __restrict__ ywarm,
    const float* __restrict__ vwarm, float* __restrict__ out) {
    extern __shared__ float sm[];
    float* v_s = sm;                 // [NN]
    float* z_s = sm + NN;            // [NN]
    float* gv  = sm + 2 * NN;        // [NN]
    float* gz  = sm + 3 * NN;        // [NN]
    float* wp  = sm + 4 * NN;        // [NN] (pbal -> w_p)
    float* wq  = sm + 5 * NN;        // [NN]
    float* red = sm + 6 * NN;        // [2]: sum_y, sum_z

    const int t = threadIdx.x;
    const int lane = t & 31;
    const float MV = mv_ptr[0];

    // Per-thread owned edge state (registers)
    float pS[EPT], qS[EPT], iS[EPT], yS[EPT];
#pragma unroll
    for (int i = 0; i < EPT; ++i) {
        int e = t + i * NT;
        pS[i] = 0.f; qS[i] = 0.f; iS[i] = 0.f;
        yS[i] = fminf(fmaxf(ywarm[e], 0.f), 1.f);
    }
#pragma unroll
    for (int i = 0; i < NPT; ++i) {
        int n = t + i * NT;
        float v0 = fminf(fmaxf(vwarm[n], 0.f), 1.3f);
        if (sub[n] > 0.f) v0 = 1.f;
        v_s[n] = v0;
        z_s[n] = 1.f;
    }
    __syncthreads();

    for (int it = 0; it < NITERS; ++it) {
        // ---- phase 0: zero accumulators
#pragma unroll
        for (int i = 0; i < NPT; ++i) {
            int n = t + i * NT;
            gv[n] = 0.f; gz[n] = 0.f; wp[n] = 0.f; wq[n] = 0.f;
        }
        if (t == 0) { red[0] = 0.f; red[1] = 0.f; }
        __syncthreads();

        // ---- phase 1: power balances (scatter) + global sums
        float sy = 0.f, sz = 0.f;
#pragma unroll
        for (int i = 0; i < EPT; ++i) {
            int e = t + i * NT;
            int s0 = src[e], d0 = dst[e];
            atomicAdd(&wp[s0],  pS[i]);
            atomicAdd(&wp[d0], -pS[i]);
            atomicAdd(&wq[s0],  qS[i]);
            atomicAdd(&wq[d0], -qS[i]);
            sy += yS[i];
        }
#pragma unroll
        for (int i = 0; i < NPT; ++i) sz += z_s[t + i * NT];
#pragma unroll
        for (int o = 16; o > 0; o >>= 1) {
            sy += __shfl_down(sy, o, 32);
            sz += __shfl_down(sz, o, 32);
        }
        if (lane == 0) { atomicAdd(&red[0], sy); atomicAdd(&red[1], sz); }
        __syncthreads();
        const float S = red[0] - red[1] + 1.f;

        // ---- phase 2: node-local gradient terms; finalize w_p, w_q
#pragma unroll
        for (int i = 0; i < NPT; ++i) {
            int n = t + i * NT;
            float v = v_s[n], z = z_s[n];
            float pj = p_inj[n], qj = q_inj[n];
            float sb = sub[n], nb = nsub[n], vw = vwarm[n];
            float wpn = 2.f * RHO_C * nb * (wp[n] - pj * z);
            float wqn = 2.f * RHO_C * nb * (wq[n] - qj * z);
            wp[n] = wpn; wq[n] = wqn;
            float gvl = 2e-3f * (v - vw) + 2.f * RHO_C * sb * (v - 1.f);
            float gzl = 2.f * RHO_C * sb * (z - 1.f) - wpn * pj - wqn * qj
                        - 2.f * RHO_C * S;
            float u3 = nb * fmaxf(0.81f * z - v, 0.f);
            gvl -= 2.f * RHO_C * u3;
            gzl += 2.f * RHO_C * 0.81f * u3;
            float u4 = nb * fmaxf(v - (1.21f + MV * (1.f - z)), 0.f);
            gvl += 2.f * RHO_C * u4;
            gzl += 2.f * RHO_C * MV * u4;
            gv[n] = gvl;
            gz[n] = gzl;
        }
        __syncthreads();

        // ---- phase 3: edge gradients; update p,q,I,y; scatter gv,gz
#pragma unroll
        for (int i = 0; i < EPT; ++i) {
            int e = t + i * NT;
            int s0 = src[e], d0 = dst[e];
            float rE = r_pu[e], xE = x_pu[e], mf = mflow[e], mq = mfsq[e];
            float zl = zlsq[e], yz = y0[e], yw = ywarm[e];
            float p = pS[i], q = qS[i], I = iS[i], y = yS[i];
            float vf = v_s[s0], vt = v_s[d0], zs = z_s[s0], zd = z_s[d0];
            float u5 = fmaxf(y - zs, 0.f);
            float u6 = fmaxf(y - zd, 0.f);
            float u7 = fmaxf(fabsf(p) - mf * y, 0.f);
            float u8 = fmaxf(fabsf(q) - mf * y, 0.f);
            float u9 = fmaxf(I - mq * y, 0.f);
            float dvi = vf - I;
            float soc = sqrtf(4.f * p * p + 4.f * q * q + dvi * dvi + EPS_C);
            float u10 = fmaxf(soc - (vf + I), 0.f);
            float vd  = vt - vf + 2.f * (rE * p + xE * q) - zl * I;
            float u11 = fmaxf(fabsf(vd) - MV * (1.f - y), 0.f);
            float svd = sgnf(vd);
            float invs = 1.f / soc;

            float gy = 2e-3f * (y - yw) + 1e-4f * sgnf(y - yz)
                     + 2.f * RHO_C * (u5 + u6 - (u7 + u8) * mf - u9 * mq
                                      + u11 * MV + S);
            float gp = 2.f * RHO_C * (u7 * sgnf(p) + u10 * 4.f * p * invs
                                      + u11 * svd * 2.f * rE)
                     + (wp[s0] - wp[d0]);
            float gq = 2.f * RHO_C * (u8 * sgnf(q) + u10 * 4.f * q * invs
                                      + u11 * svd * 2.f * xE)
                     + (wq[s0] - wq[d0]);
            float gI = rE + 2.f * RHO_C * (u9 + u10 * (-dvi * invs - 1.f)
                                           - u11 * svd * zl);
            float gvs = 2.f * RHO_C * (u10 * (dvi * invs - 1.f) - u11 * svd);
            float gvd = 2.f * RHO_C * (u11 * svd);
            atomicAdd(&gv[s0], gvs);
            atomicAdd(&gv[d0], gvd);
            atomicAdd(&gz[s0], -2.f * RHO_C * u5);
            atomicAdd(&gz[d0], -2.f * RHO_C * u6);

            pS[i] = p - LR_C * gp;
            qS[i] = q - LR_C * gq;
            iS[i] = fmaxf(I - LR_C * gI, 0.f);
            yS[i] = fminf(fmaxf(y - LR_C * gy, 0.f), 1.f);
        }
        __syncthreads();

        // ---- phase 4: node updates + projection
#pragma unroll
        for (int i = 0; i < NPT; ++i) {
            int n = t + i * NT;
            float sb = sub[n];
            float v = v_s[n] - LR_C * gv[n];
            v = fminf(fmaxf(v, 0.f), 1.3f);
            if (sb > 0.f) v = 1.f;
            float z = z_s[n] - LR_C * gz[n];
            z = fminf(fmaxf(z, 0.f), 1.f);
            if (sb > 0.f) z = 1.f;
            v_s[n] = v;
            z_s[n] = z;
        }
        __syncthreads();
    }

    // outputs: y_opt [EE] then v_sq_opt [NN]
#pragma unroll
    for (int i = 0; i < EPT; ++i) out[t + i * NT] = yS[i];
#pragma unroll
    for (int i = 0; i < NPT; ++i) out[EE + t + i * NT] = v_s[t + i * NT];
}

// ---------------------------------------------------------------------------
// Host launcher
// ---------------------------------------------------------------------------
extern "C" void kernel_launch(void* const* d_in, const int* in_sizes, int n_in,
                              void* d_out, int out_size, void* d_ws, size_t ws_size,
                              hipStream_t stream) {
    const float* x        = (const float*)d_in[0];
    const int*   eidx     = (const int*)d_in[1];
    const float* p_inj    = (const float*)d_in[2];
    const float* q_inj    = (const float*)d_in[3];
    const float* y0       = (const float*)d_in[4];
    const float* r_pu     = (const float*)d_in[5];
    const float* x_pu     = (const float*)d_in[6];
    const float* bigMf    = (const float*)d_in[7];
    const float* bigMv    = (const float*)d_in[8];
    // d_in[9], d_in[10]: dense A_from/A_to — intentionally unused (incidence == edge_index)
    const float* sub      = (const float*)d_in[11];
    const float* nsub     = (const float*)d_in[12];
    const float* bigMfsq  = (const float*)d_in[13];
    const float* zlsq     = (const float*)d_in[14];
    const float* W_enc    = (const float*)d_in[15];
    const float* b_enc    = (const float*)d_in[16];
    const float* W_g1     = (const float*)d_in[17];
    const float* b_g1     = (const float*)d_in[18];
    const float* W_g2     = (const float*)d_in[19];
    const float* b_g2     = (const float*)d_in[20];
    const float* W_sw     = (const float*)d_in[21];
    const float* b_sw     = (const float*)d_in[22];
    const float* W_v      = (const float*)d_in[23];
    const float* b_v      = (const float*)d_in[24];
    float* out = (float*)d_out;

    const int* src = eidx;
    const int* dst = eidx + EE;

    // ---- workspace layout (bump allocation, 256B-aligned pieces)
    char* ws = (char*)d_ws;
    size_t off = 0;
    auto alloc = [&](size_t bytes) {
        char* p = ws + off;
        off += (bytes + 255) & ~(size_t)255;
        return p;
    };
    float*    h32   = (float*)   alloc((size_t)NN * HF * 4);
    float*    tbuf  = (float*)   alloc((size_t)NN * HF * 4);
    float*    agg   = (float*)   alloc((size_t)NN * HF * 4);
    _Float16* h16   = (_Float16*)alloc((size_t)NN * HF * 2);
    _Float16* xh    = (_Float16*)alloc((size_t)NN * DIN * 2);
    _Float16* wencT = (_Float16*)alloc((size_t)DIN * HF * 2);
    _Float16* wg1T  = (_Float16*)alloc((size_t)HF * HF * 2);
    _Float16* wg2T  = (_Float16*)alloc((size_t)HF * HF * 2);
    float*    deg   = (float*)   alloc((size_t)NN * 4);
    float*    dis   = (float*)   alloc((size_t)NN * 4);
    float*    s1    = (float*)   alloc((size_t)NN * 4);
    float*    s2    = (float*)   alloc((size_t)NN * 4);
    float*    vwarm = (float*)   alloc((size_t)NN * 4);
    float*    ywarm = (float*)   alloc((size_t)EE * 4);
    (void)ws_size; (void)n_in; (void)in_sizes; (void)out_size;

    const int TPB = 256;
    // ---- precompute: f16 conversions + degree normalization
    k_f32_to_f16<<<(NN * DIN + TPB - 1) / TPB, TPB, 0, stream>>>(x, xh, NN * DIN);
    k_transpose_f16<<<(DIN * HF + TPB - 1) / TPB, TPB, 0, stream>>>(W_enc, wencT, DIN, HF);
    k_transpose_f16<<<(HF * HF + TPB - 1) / TPB, TPB, 0, stream>>>(W_g1, wg1T, HF, HF);
    k_transpose_f16<<<(HF * HF + TPB - 1) / TPB, TPB, 0, stream>>>(W_g2, wg2T, HF, HF);
    k_zero<<<NN / TPB, TPB, 0, stream>>>(deg, NN);
    k_deg_scatter<<<EE / TPB, TPB, 0, stream>>>(dst, deg, EE);
    k_deg_fin<<<NN / TPB, TPB, 0, stream>>>(deg, dis, NN);

    const int gemmBlocks = (NN / 16) * (HF / 16) / 4;  // 4 waves per block

    // ---- encoder: h = relu(x @ W_enc + b_enc)
    k_gemm_f16<<<gemmBlocks, 128, 0, stream>>>(xh, wencT, tbuf, NN, HF, DIN);
    k_finalize<<<(NN * HF) / TPB, TPB, 0, stream>>>(tbuf, b_enc, h32, h16, NN * HF);

    // ---- GCN layer 1
    k_gemm_f16<<<gemmBlocks, 128, 0, stream>>>(h16, wg1T, tbuf, NN, HF, HF);
    k_agg_init<<<(NN * HF) / TPB, TPB, 0, stream>>>(tbuf, dis, agg, NN * HF);
    k_edge_scatter<<<EE, HF, 0, stream>>>(src, dst, dis, tbuf, agg);
    k_finalize<<<(NN * HF) / TPB, TPB, 0, stream>>>(agg, b_g1, h32, h16, NN * HF);

    // ---- GCN layer 2
    k_gemm_f16<<<gemmBlocks, 128, 0, stream>>>(h16, wg2T, tbuf, NN, HF, HF);
    k_agg_init<<<(NN * HF) / TPB, TPB, 0, stream>>>(tbuf, dis, agg, NN * HF);
    k_edge_scatter<<<EE, HF, 0, stream>>>(src, dst, dis, tbuf, agg);
    k_finalize<<<(NN * HF) / TPB, TPB, 0, stream>>>(agg, b_g2, h32, h16, NN * HF);

    // ---- heads: warm starts
    k_heads<<<NN / 8, 256, 0, stream>>>(h32, W_sw, W_v, b_v, s1, s2, vwarm);
    k_ywarm<<<EE / TPB, TPB, 0, stream>>>(src, dst, s1, s2, b_sw, ywarm, EE);

    // ---- persistent single-WGP penalty solve (state in LDS + registers)
    size_t smem = (size_t)(6 * NN + 32) * sizeof(float);  // ~96.1 KB of 320 KB LDS
    k_solve<<<1, NT, smem, stream>>>(src, dst, p_inj, q_inj, y0, r_pu, x_pu,
                                     bigMf, bigMv, sub, nsub, bigMfsq, zlsq,
                                     ywarm, vwarm, out);
}
